// Vim_30288109371735
// MI455X (gfx1250) — compile-verified, matrix operand order checked by reading, not run on hardware
//
#include <hip/hip_runtime.h>
#include <math.h>

#define DIM 768
#define DT_RANK 32
#define D_STATE 16
#define DEPTH 12
#define NCLS 1000
#define BATCH 16
#define SEQ 197
#define NPATCH 196
#define TOK (BATCH * SEQ)      /* 3152 */
#define PTOK (BATCH * NPATCH)  /* 3136 */

typedef __attribute__((ext_vector_type(16))) __bf16 v16bf;
typedef __attribute__((ext_vector_type(8)))  __bf16 v8bf;
typedef __attribute__((ext_vector_type(8)))  float  v8f;

__device__ __forceinline__ float softplus_f(float x) {
  return (x > 20.f) ? x : log1pf(expf(x));
}

__device__ __forceinline__ v8bf pack8(float4 f0, float4 f1) {
  v8bf p;
  p[0] = (__bf16)f0.x; p[1] = (__bf16)f0.y; p[2] = (__bf16)f0.z; p[3] = (__bf16)f0.w;
  p[4] = (__bf16)f1.x; p[5] = (__bf16)f1.y; p[6] = (__bf16)f1.z; p[7] = (__bf16)f1.w;
  return p;
}

__device__ __forceinline__ v16bf cat16(v8bf lo, v8bf hi) {
  return __builtin_shufflevector(lo, hi, 0, 1, 2, 3, 4, 5, 6, 7,
                                         8, 9, 10, 11, 12, 13, 14, 15);
}

// ---------------------------------------------------------------------------
// C[M,N] = act( A[M,K] @ B + bias ), bf16 WMMA, f32 accumulate.
// Block tile 128x64, BK=32, 256 threads = 8 waves. Wave w owns m-subtile w and
// all 4 n-subtiles -> one A fragment feeds 4x v_wmma_f32_16x16x32_bf16.
// TRANSB==0: B is (K,N) ldb.  TRANSB==1: B is (N,K) ldb.  ACT: 0 none, 1 softplus.
// ---------------------------------------------------------------------------
template <int TRANSB, int ACT>
__global__ __launch_bounds__(256) void gemm_wmma_kernel(
    const float* __restrict__ A, const float* __restrict__ Bg,
    float* __restrict__ C, const float* __restrict__ bias,
    int M, int N, int K, int lda, int ldb, int ldc)
{
  __shared__ __bf16 As[128 * 32];  // [m][k], k contiguous
  __shared__ __bf16 Bs[64 * 32];   // [n][k], k contiguous

  const int tid  = threadIdx.x;
  const int lane = tid & 31;
  const int wave = tid >> 5;               // == m-subtile
  const int m0 = blockIdx.y * 128;
  const int n0 = blockIdx.x * 64;

  v8f acc[4] = {v8f{}, v8f{}, v8f{}, v8f{}};

  const int mn = lane & 15;
  const int kg = (lane >> 4) << 3;         // 0 or 8

  // staging coordinates (uniform per thread across K loop)
  const int am = tid >> 1;                 // 0..127
  const int ak = (tid & 1) * 16;           // 0 or 16
  const int agm = min(m0 + am, M - 1);
  const int bn  = tid >> 2;                // 0..63   (TRANSB==1)
  const int bk1 = (tid & 3) * 8;
  const int bgn = min(n0 + bn, N - 1);
  const int bk0 = tid >> 3;                // 0..31   (TRANSB==0)
  const int bng = (tid & 7) * 8;

  for (int k0 = 0; k0 < K; k0 += 32) {
    __syncthreads();
    // ---- stage A tile (4x float4 per thread -> 2x b128 LDS store) ----
    {
      const float* ap = A + (size_t)agm * lda + k0 + ak;
      if (k0 + 32 < K) __builtin_prefetch(ap + 32, 0, 0);
      float4 f0 = *(const float4*)(ap);
      float4 f1 = *(const float4*)(ap + 4);
      float4 f2 = *(const float4*)(ap + 8);
      float4 f3 = *(const float4*)(ap + 12);
      *(v8bf*)(&As[am * 32 + ak])     = pack8(f0, f1);
      *(v8bf*)(&As[am * 32 + ak + 8]) = pack8(f2, f3);
    }
    // ---- stage B tile into [n][k] ----
    if (TRANSB) {
      const float* bp = Bg + (size_t)bgn * ldb + k0 + bk1;
      float4 f0 = *(const float4*)(bp);
      float4 f1 = *(const float4*)(bp + 4);
      *(v8bf*)(&Bs[bn * 32 + bk1]) = pack8(f0, f1);
    } else {
      const float* bp = Bg + (size_t)(k0 + bk0) * ldb + n0 + bng;
      float vals[8];
      if (((ldb & 3) == 0) && (n0 + bng + 7 < N)) {
        float4 f0 = *(const float4*)(bp);
        float4 f1 = *(const float4*)(bp + 4);
        vals[0] = f0.x; vals[1] = f0.y; vals[2] = f0.z; vals[3] = f0.w;
        vals[4] = f1.x; vals[5] = f1.y; vals[6] = f1.z; vals[7] = f1.w;
      } else {
#pragma unroll
        for (int j = 0; j < 8; ++j)
          vals[j] = (n0 + bng + j < N) ? bp[j] : 0.f;
      }
#pragma unroll
      for (int j = 0; j < 8; ++j)
        Bs[(bng + j) * 32 + bk0] = (__bf16)vals[j];
    }
    __syncthreads();

    // ---- fragments + 4 WMMAs ----
    const __bf16* arow = &As[((wave << 4) + mn) * 32];
    v16bf afrag = cat16(*(const v8bf*)(arow + kg), *(const v8bf*)(arow + kg + 16));
#pragma unroll
    for (int nt = 0; nt < 4; ++nt) {
      const __bf16* brow = &Bs[((nt << 4) + mn) * 32];
      v16bf bfrag = cat16(*(const v8bf*)(brow + kg), *(const v8bf*)(brow + kg + 16));
      acc[nt] = __builtin_amdgcn_wmma_f32_16x16x32_bf16(
          false, afrag, false, bfrag, (short)0, acc[nt], false, false);
    }
  }

  // ---- epilogue: D layout VGPR r -> M=r (+8 for lanes 16-31), N=lane&15 ----
  const int mrow = (lane >> 4) << 3;
  const int ncol = lane & 15;
#pragma unroll
  for (int nt = 0; nt < 4; ++nt) {
    int n = n0 + (nt << 4) + ncol;
    float bv = (bias && n < N) ? bias[n] : 0.f;
#pragma unroll
    for (int r = 0; r < 8; ++r) {
      int m = m0 + (wave << 4) + mrow + r;
      if (m < M && n < N) {
        float v = acc[nt][r] + bv;
        if (ACT == 1) v = softplus_f(v);
        C[(size_t)m * ldc + n] = v;
      }
    }
  }
}

// ---------------------------------------------------------------------------
__global__ void patchify_kernel(const float* __restrict__ x, float* __restrict__ xp)
{
  int idx = blockIdx.x * blockDim.x + threadIdx.x;
  if (idx >= PTOK * DIM) return;
  int row = idx / DIM, col = idx % DIM;
  int b = row / NPATCH, s = row % NPATCH;
  int hh = s / 14, ww = s % 14;
  int p1 = col / 48, r = col % 48;
  int p2 = r / 3, c = r % 3;
  xp[idx] = x[(((size_t)b * 3 + c) * 224 + hh * 16 + p1) * 224 + ww * 16 + p2];
}

__global__ void assemble_kernel(const float* __restrict__ xp_out,
                                const float* __restrict__ cls,
                                float* __restrict__ h)
{
  int idx = blockIdx.x * blockDim.x + threadIdx.x;
  if (idx >= TOK * DIM) return;
  int row = idx / DIM, d = idx % DIM;
  int b = row / SEQ, s = row % SEQ;
  h[idx] = (s == 0) ? cls[d]
                    : xp_out[((size_t)b * NPATCH + (s - 1)) * DIM + d];
}

// ---------------------------------------------------------------------------
__global__ __launch_bounds__(256) void layernorm_kernel(
    const float* __restrict__ in, float* __restrict__ out,
    const float* __restrict__ g, const float* __restrict__ b)
{
  __shared__ float red[256];
  int row = blockIdx.x;
  const float* xr = in + (size_t)row * DIM;

  float s = 0.f;
  for (int d = threadIdx.x; d < DIM; d += 256) s += xr[d];
  red[threadIdx.x] = s; __syncthreads();
  for (int off = 128; off > 0; off >>= 1) {
    if (threadIdx.x < off) red[threadIdx.x] += red[threadIdx.x + off];
    __syncthreads();
  }
  float mu = red[0] / DIM;
  __syncthreads();

  float v = 0.f;
  for (int d = threadIdx.x; d < DIM; d += 256) {
    float t = xr[d] - mu; v += t * t;
  }
  red[threadIdx.x] = v; __syncthreads();
  for (int off = 128; off > 0; off >>= 1) {
    if (threadIdx.x < off) red[threadIdx.x] += red[threadIdx.x + off];
    __syncthreads();
  }
  float rs = rsqrtf(red[0] / DIM + 1e-5f);

  for (int d = threadIdx.x; d < DIM; d += 256)
    out[(size_t)row * DIM + d] = (xr[d] - mu) * rs * g[d] + b[d];
}

// float4 elementwise kernels (counts are multiples of 4: DIM=768)
__global__ void silu4_kernel(const float4* __restrict__ in, float4* __restrict__ out, int n4)
{
  int i = blockIdx.x * blockDim.x + threadIdx.x;
  if (i >= n4) return;
  float4 v = in[i];
  v.x = v.x / (1.f + expf(-v.x));
  v.y = v.y / (1.f + expf(-v.y));
  v.z = v.z / (1.f + expf(-v.z));
  v.w = v.w / (1.f + expf(-v.w));
  out[i] = v;
}

__global__ void residual4_kernel(float4* __restrict__ h,
                                 const float4* __restrict__ x1,
                                 const float4* __restrict__ x2,
                                 const float4* __restrict__ z, int n4)
{
  int i = blockIdx.x * blockDim.x + threadIdx.x;
  if (i >= n4) return;
  float4 a = x1[i], b = x2[i], zz = z[i], hv = h[i];
  hv.x += (a.x + b.x) * zz.x;
  hv.y += (a.y + b.y) * zz.y;
  hv.z += (a.z + b.z) * zz.z;
  hv.w += (a.w + b.w) * zz.w;
  h[i] = hv;
}

// ---------------------------------------------------------------------------
// Selective scan: lanes 0-15 -> channel e0 (n=lane), lanes 16-31 -> e0+1.
// ---------------------------------------------------------------------------
__global__ __launch_bounds__(256) void scan_kernel(
    const float* __restrict__ u, const float* __restrict__ delta,
    const float* __restrict__ dbc, const float* __restrict__ A_log,
    const float* __restrict__ Dp, float* __restrict__ y)
{
  int wave = threadIdx.x >> 5;
  int lane = threadIdx.x & 31;
  int n = lane & 15;
  int e = (blockIdx.x * 8 + wave) * 2 + (lane >> 4);
  int b = blockIdx.y;

  float Aen = -expf(A_log[(size_t)e * D_STATE + n]);
  float De  = Dp[e];
  float hs  = 0.f;

  for (int l = 0; l < SEQ; ++l) {
    size_t tok = (size_t)b * SEQ + l;
    float dlt = delta[tok * DIM + e];
    float uu  = u[tok * DIM + e];
    float Bn  = dbc[tok * 64 + DT_RANK + n];
    float Cn  = dbc[tok * 64 + DT_RANK + D_STATE + n];
    float dA  = expf(dlt * Aen);
    hs = hs * dA + dlt * Bn * uu;
    float part = hs * Cn;
    part += __shfl_xor(part, 1, 32);
    part += __shfl_xor(part, 2, 32);
    part += __shfl_xor(part, 4, 32);
    part += __shfl_xor(part, 8, 32);
    if (n == 0) y[tok * DIM + e] = part + De * uu;
  }
}

__global__ void pool_kernel(const float* __restrict__ h, float* __restrict__ pooled)
{
  int b = blockIdx.x, d = threadIdx.x;   // block = DIM threads
  float s = 0.f;
  for (int l = 0; l < SEQ; ++l) s += h[((size_t)b * SEQ + l) * DIM + d];
  pooled[(size_t)b * DIM + d] = s / (float)SEQ;
}

// ---------------------------------------------------------------------------
extern "C" void kernel_launch(void* const* d_in, const int* in_sizes, int n_in,
                              void* d_out, int out_size, void* d_ws, size_t ws_size,
                              hipStream_t stream)
{
  const float* x        = (const float*)d_in[0];
  const float* patch_W  = (const float*)d_in[1];
  const float* patch_b  = (const float*)d_in[2];
  const float* cls_tok  = (const float*)d_in[3];
  const float* norm_g   = (const float*)d_in[4];
  const float* norm_b   = (const float*)d_in[5];
  const float* proj_W   = (const float*)d_in[6];
  const float* proj_b   = (const float*)d_in[7];
  const float* fconv_W  = (const float*)d_in[8];
  const float* fconv_b  = (const float*)d_in[9];
  const float* bconv_W  = (const float*)d_in[10];
  const float* bconv_b  = (const float*)d_in[11];
  const float* dbc_W    = (const float*)d_in[12];
  const float* dt_W     = (const float*)d_in[13];
  const float* dt_b     = (const float*)d_in[14];
  const float* A_log    = (const float*)d_in[15];
  const float* D_ssm    = (const float*)d_in[16];
  const float* head_g   = (const float*)d_in[17];
  const float* head_b   = (const float*)d_in[18];
  const float* head_W   = (const float*)d_in[19];
  const float* head_bias= (const float*)d_in[20];
  float* out = (float*)d_out;

  float* ws = (float*)d_ws;
  size_t o = 0;
  auto alloc = [&](size_t n) { float* p = ws + o; o += n; return p; };
  float* xp_mat = alloc((size_t)PTOK * DIM);
  float* h      = alloc((size_t)TOK * DIM);
  float* xn     = alloc((size_t)TOK * DIM);
  float* xz     = alloc((size_t)TOK * DIM);
  float* z      = alloc((size_t)TOK * DIM);
  float* u      = alloc((size_t)TOK * DIM);   // also reused as xp_out
  float* delta  = alloc((size_t)TOK * DIM);
  float* x1     = alloc((size_t)TOK * DIM);
  float* x2     = alloc((size_t)TOK * DIM);
  float* dbc    = alloc((size_t)TOK * 64);
  float* pooled = alloc((size_t)BATCH * DIM);
  float* pooln  = alloc((size_t)BATCH * DIM);
  (void)ws_size; (void)in_sizes; (void)n_in; (void)out_size;

  const int EW = 256;
  const int N4 = TOK * DIM / 4;
  dim3 g4((N4 + EW - 1) / EW);

  // --- patch embed -------------------------------------------------------
  patchify_kernel<<<(PTOK * DIM + EW - 1) / EW, EW, 0, stream>>>(x, xp_mat);
  gemm_wmma_kernel<0, 0><<<dim3((DIM + 63) / 64, (PTOK + 127) / 128), 256, 0, stream>>>(
      xp_mat, patch_W, u, patch_b, PTOK, DIM, DIM, DIM, DIM, DIM);
  assemble_kernel<<<(TOK * DIM + EW - 1) / EW, EW, 0, stream>>>(u, cls_tok, h);

  // --- layers ------------------------------------------------------------
  dim3 gM((DIM + 63) / 64, (TOK + 127) / 128);
  dim3 gDbc(1, (TOK + 127) / 128);
  for (int i = 0; i < DEPTH; ++i) {
    const float* projW_i = proj_W + (size_t)i * DIM * DIM;
    const float* projb_i = proj_b + (size_t)i * DIM;
    const float* dbcW_i  = dbc_W  + (size_t)i * DIM * (DT_RANK + 2 * D_STATE);
    const float* dtW_i   = dt_W   + (size_t)i * DT_RANK * DIM;
    const float* dtb_i   = dt_b   + (size_t)i * DIM;
    const float* Alog_i  = A_log  + (size_t)i * DIM * D_STATE;
    const float* D_i     = D_ssm  + (size_t)i * DIM;

    layernorm_kernel<<<TOK, 256, 0, stream>>>(h, xn,
        norm_g + (size_t)i * DIM, norm_b + (size_t)i * DIM);

    gemm_wmma_kernel<0, 0><<<gM, 256, 0, stream>>>(
        xn, projW_i, xz, projb_i, TOK, DIM, DIM, DIM, DIM, DIM);
    silu4_kernel<<<g4, EW, 0, stream>>>((const float4*)xz, (float4*)z, N4);

    for (int dir = 0; dir < 2; ++dir) {
      const float* cW = (dir == 0) ? fconv_W + (size_t)i * DIM * DIM
                                   : bconv_W + (size_t)i * DIM * DIM;
      const float* cb = (dir == 0) ? fconv_b + (size_t)i * DIM
                                   : bconv_b + (size_t)i * DIM;
      float* ydir = (dir == 0) ? x1 : x2;

      // u = softplus(xz @ cW^T + cb)
      gemm_wmma_kernel<1, 1><<<gM, 256, 0, stream>>>(
          xz, cW, u, cb, TOK, DIM, DIM, DIM, DIM, DIM);
      // dbc = u @ dbc_W  (N=64)
      gemm_wmma_kernel<0, 0><<<gDbc, 256, 0, stream>>>(
          u, dbcW_i, dbc, nullptr, TOK, 64, DIM, DIM, 64, 64);
      // delta = softplus(dbc[:, :32] @ dt_W + dt_b)  (K=32, lda=64)
      gemm_wmma_kernel<0, 1><<<gM, 256, 0, stream>>>(
          dbc, dtW_i, delta, dtb_i, TOK, DIM, DT_RANK, 64, DIM, DIM);

      scan_kernel<<<dim3(DIM / 16, BATCH), 256, 0, stream>>>(
          u, delta, dbc, Alog_i, D_i, ydir);
    }

    residual4_kernel<<<g4, EW, 0, stream>>>(
        (float4*)h, (const float4*)x1, (const float4*)x2, (const float4*)z, N4);
  }

  // --- head --------------------------------------------------------------
  pool_kernel<<<BATCH, DIM, 0, stream>>>(h, pooled);
  layernorm_kernel<<<BATCH, 256, 0, stream>>>(pooled, pooln, head_g, head_b);
  gemm_wmma_kernel<0, 0><<<dim3((NCLS + 63) / 64, 1), 256, 0, stream>>>(
      pooln, head_W, out, head_bias, BATCH, NCLS, DIM, DIM, NCLS, NCLS);
}
